// GraphLayer_88132728914046
// MI455X (gfx1250) — matrix operand order repaired and checked by reading, not compiled, not run on hardware
//
#include <hip/hip_runtime.h>
#include <math.h>

// ---------------------------------------------------------------------------
// GraphGRU layer for MI455X (gfx1250, wave32, WMMA bf16 16x16x32).
//
// Pipeline:
//   k_prep      : weights -> bf16, transposed  (ws)
//   k_encode    : x_enc = x @ W_encode + b      (fp32 ws + bf16-transposed ws)
//   k_zero      : zero the aggregation partial buffer
//   k_aggregate : a = adj @ x_enc  (HBM-bound: 268 MB adj streamed once;
//                 K split 4 ways -> ~4 waves/SIMD; global_atomic_add_f32)
//   k_gate      : z/r/h gating, 6 fused 64x64 GEMMs + LDS relayout for r*x
//
// Workspace layout (needs ~10.6 MB):
//   [0]        WencT  bf16 [64][128]            16384 B
//   [16384]    WT     bf16 6 x [64][64]         49152 B
//   [65536]    xT     bf16 [64][B*N]          2097152 B
//   [2162688]  xe     f32  [B*N][64]          4194304 B
//   [6356992]  a_ws   f32  [B*N][64]          4194304 B
// ---------------------------------------------------------------------------

typedef __attribute__((ext_vector_type(16))) __bf16 v16bf;
typedef __attribute__((ext_vector_type(8)))  float  v8f;

#define B_SZ   4
#define N_SZ   4096
#define DIN    128
#define DOUT   64
#define ROWS   (B_SZ * N_SZ)
#define KSPLIT 4                      // K-slices in k_aggregate (occupancy)

// ---- WMMA wrapper ----------------------------------------------------------
__device__ __forceinline__ v8f wmma_bf16(v16bf a, v16bf b, v8f c) {
  // (neg_a, A, neg_b, B, c_mod, C, reuse_a, reuse_b)
  return __builtin_amdgcn_wmma_f32_16x16x32_bf16(false, a, false, b,
                                                 (short)0, c, false, false);
}

// ---- A-fragment (16x32, MxK) from row-major fp32, converting to bf16 -------
// Layout (ISA 7.12.2): lane l holds row m = l&15; lane-half hi = l>>4.
//   halves 0..7  -> K = kk + 8*hi + i          (contiguous 8 floats)
//   halves 8..15 -> K = kk + 16 + 8*hi + (i-8) (contiguous 8 floats)
__device__ __forceinline__ v16bf make_afrag_f32(const float* __restrict__ base,
                                                int ld, int m, int hi, int kk) {
  const float* p = base + (size_t)m * ld + kk + 8 * hi;
  const float4 f0 = *(const float4*)(p + 0);
  const float4 f1 = *(const float4*)(p + 4);
  const float4 f2 = *(const float4*)(p + 16);
  const float4 f3 = *(const float4*)(p + 20);
  v16bf a;
  a[0]  = (__bf16)f0.x; a[1]  = (__bf16)f0.y; a[2]  = (__bf16)f0.z; a[3]  = (__bf16)f0.w;
  a[4]  = (__bf16)f1.x; a[5]  = (__bf16)f1.y; a[6]  = (__bf16)f1.z; a[7]  = (__bf16)f1.w;
  a[8]  = (__bf16)f2.x; a[9]  = (__bf16)f2.y; a[10] = (__bf16)f2.z; a[11] = (__bf16)f2.w;
  a[12] = (__bf16)f3.x; a[13] = (__bf16)f3.y; a[14] = (__bf16)f3.z; a[15] = (__bf16)f3.w;
  return a;
}

// ---- B-fragment (32x16, KxN) from K-contiguous transposed bf16 -------------
// Layout: lane l holds col n = l&15; halves i<8 -> K = kk + 8*hi + i,
// halves i>=8 -> K = kk + 16 + 8*hi + (i-8) (two 16-byte loads per lane).
__device__ __forceinline__ v16bf ld_bfrag(const __bf16* __restrict__ wt,
                                          int rowlen, int n, int hi, int kk) {
  typedef __attribute__((ext_vector_type(8))) __bf16 v8bf;
  const __bf16* p = wt + (size_t)n * rowlen + kk + 8 * hi;
  const v8bf lo = *(const v8bf*)(p + 0);
  const v8bf up = *(const v8bf*)(p + 16);
  return __builtin_shufflevector(lo, up, 0, 1, 2, 3, 4, 5, 6, 7, 8, 9, 10, 11, 12, 13, 14, 15);
}

__device__ __forceinline__ float sigmoidf(float v) {
  return 1.0f / (1.0f + __expf(-v));
}

// ---------------------------------------------------------------------------
// k_prep: convert weights to bf16, transposed (row = output col n, K-contig).
// ---------------------------------------------------------------------------
__global__ void k_prep(const float* __restrict__ We,
                       const float* __restrict__ W0, const float* __restrict__ W1,
                       const float* __restrict__ W2, const float* __restrict__ W3,
                       const float* __restrict__ W4, const float* __restrict__ W5,
                       __bf16* __restrict__ WeT, __bf16* __restrict__ WT) {
  const int t = blockIdx.x * blockDim.x + threadIdx.x;
  const int stride = gridDim.x * blockDim.x;
  for (int i = t; i < DOUT * DIN; i += stride) {
    int n = i / DIN, k = i % DIN;
    WeT[i] = (__bf16)We[k * DOUT + n];
  }
  const float* Ws[6] = {W0, W1, W2, W3, W4, W5};
  for (int w = 0; w < 6; ++w)
    for (int i = t; i < DOUT * DOUT; i += stride) {
      int n = i / DOUT, k = i % DOUT;
      WT[w * DOUT * DOUT + i] = (__bf16)Ws[w][k * DOUT + n];
    }
}

// ---------------------------------------------------------------------------
// k_zero: zero the aggregation partial-sum buffer (graph-capture-safe).
// ---------------------------------------------------------------------------
__global__ void k_zero(float4* __restrict__ p, int n4) {
  const int t = blockIdx.x * blockDim.x + threadIdx.x;
  if (t < n4) p[t] = make_float4(0.f, 0.f, 0.f, 0.f);
}

// ---------------------------------------------------------------------------
// k_encode: xe = x @ W_encode + b_enc ; also store xe^T as bf16 for stage 2.
// 128 threads = 4 waves, each wave one 16-row strip x 64 cols (4 WMMA tiles).
// ---------------------------------------------------------------------------
__global__ void __launch_bounds__(128)
k_encode(const float* __restrict__ x, const __bf16* __restrict__ WeT,
         const float* __restrict__ b_enc,
         float* __restrict__ xe, __bf16* __restrict__ xT) {
  const int lane = threadIdx.x & 31;
  const int wave = threadIdx.x >> 5;
  const int m = lane & 15, hi = lane >> 4, nl = lane & 15;
  const int row0 = blockIdx.x * 64 + wave * 16;
  const float* base = x + (size_t)row0 * DIN;

  v8f acc[4] = {};
#pragma unroll
  for (int kk = 0; kk < DIN; kk += 32) {
    v16bf a = make_afrag_f32(base, DIN, m, hi, kk);
#pragma unroll
    for (int t = 0; t < 4; ++t)
      acc[t] = wmma_bf16(a, ld_bfrag(WeT, DIN, 16 * t + nl, hi, kk), acc[t]);
  }
#pragma unroll
  for (int t = 0; t < 4; ++t) {
    const int col = 16 * t + nl;
    const float bias = b_enc[col];
#pragma unroll
    for (int r = 0; r < 8; ++r) {
      const int row = row0 + r + 8 * hi;   // C-layout: VGPR r -> M = r + 8*hi
      const float v = acc[t][r] + bias;
      xe[(size_t)row * DOUT + col] = v;
      xT[(size_t)col * ROWS + row] = (__bf16)v;
    }
  }
}

// ---------------------------------------------------------------------------
// k_aggregate: a = adj[b] @ x_enc[b].  The HBM-bound stage: each wave streams
// a 16-row strip of adj (fp32 -> bf16 in regs) against L2-resident x_enc^T.
// K reduction split KSPLIT ways (blockIdx.z) for ~4 waves/SIMD of latency
// hiding; partials combined with hardware global_atomic_add_f32.
// grid = (N/64, B, KSPLIT), 128 threads = 4 waves.
// ---------------------------------------------------------------------------
__global__ void __launch_bounds__(128)
k_aggregate(const float* __restrict__ adj, const __bf16* __restrict__ xT,
            float* __restrict__ a_ws) {
  const int lane = threadIdx.x & 31;
  const int wave = threadIdx.x >> 5;
  const int m = lane & 15, hi = lane >> 4, nl = lane & 15;
  const int b = blockIdx.y;
  const int r0 = blockIdx.x * 64 + wave * 16;
  const int k0 = blockIdx.z * (N_SZ / KSPLIT);
  const int k1 = k0 + (N_SZ / KSPLIT);
  const float* base = adj + ((size_t)b * N_SZ + r0) * N_SZ;
  const int kbase = b * N_SZ;  // K index of this batch inside xT rows

  v8f acc[4] = {};
#pragma unroll 4
  for (int kk = k0; kk < k1; kk += 32) {
    v16bf a = make_afrag_f32(base, N_SZ, m, hi, kk);
    // CDNA5 global_prefetch_b8: pull the next chunk of this lane's adj row.
    __builtin_prefetch(base + (size_t)m * N_SZ + kk + 8 * hi + 512, 0, 1);
#pragma unroll
    for (int t = 0; t < 4; ++t)
      acc[t] = wmma_bf16(a, ld_bfrag(xT, ROWS, 16 * t + nl, hi, kbase + kk),
                         acc[t]);
  }
  float* arow = a_ws + ((size_t)b * N_SZ + r0) * DOUT;
#pragma unroll
  for (int t = 0; t < 4; ++t)
#pragma unroll
    for (int r = 0; r < 8; ++r)
      unsafeAtomicAdd(&arow[(size_t)(r + 8 * hi) * DOUT + 16 * t + nl],
                      acc[t][r]);
}

// ---------------------------------------------------------------------------
// k_gate: z = sig(a@Wz0 + x@Wz1 + bz), r = sig(a@Wr0 + x@Wr1 + br),
//         h = relu(mask * (a@Wh0 + (r*x)@Wh1 + bh)), out = h*z + x*(1-z).
// Per wave: 16 rows, 48 WMMAs. r is relayouted C->A via per-wave LDS tile.
// ---------------------------------------------------------------------------
__global__ void __launch_bounds__(128)
k_gate(const float* __restrict__ a_ws, const float* __restrict__ xe,
       const __bf16* __restrict__ WT, const float* __restrict__ mask,
       const float* __restrict__ bz0, const float* __restrict__ bz1,
       const float* __restrict__ br0, const float* __restrict__ br1,
       const float* __restrict__ bh0, const float* __restrict__ bh1,
       float* __restrict__ out) {
  __shared__ float lds_r[4][16 * 64];
  const int lane = threadIdx.x & 31;
  const int wave = threadIdx.x >> 5;
  const int m = lane & 15, hi = lane >> 4, nl = lane & 15;
  const int row0 = blockIdx.x * 64 + wave * 16;

  v16bf aA[2], xA[2];
#pragma unroll
  for (int kc = 0; kc < 2; ++kc) {
    aA[kc] = make_afrag_f32(a_ws + (size_t)row0 * DOUT, DOUT, m, hi, kc * 32);
    xA[kc] = make_afrag_f32(xe   + (size_t)row0 * DOUT, DOUT, m, hi, kc * 32);
  }
  const __bf16* Wz0T = WT;
  const __bf16* Wz1T = WT + 4096;
  const __bf16* Wr0T = WT + 8192;
  const __bf16* Wr1T = WT + 12288;
  const __bf16* Wh0T = WT + 16384;
  const __bf16* Wh1T = WT + 20480;

  v8f zac[4] = {}, rac[4] = {};
#pragma unroll
  for (int t = 0; t < 4; ++t)
#pragma unroll
    for (int kc = 0; kc < 2; ++kc) {
      const int n = 16 * t + nl, kk = kc * 32;
      zac[t] = wmma_bf16(aA[kc], ld_bfrag(Wz0T, DOUT, n, hi, kk), zac[t]);
      zac[t] = wmma_bf16(xA[kc], ld_bfrag(Wz1T, DOUT, n, hi, kk), zac[t]);
      rac[t] = wmma_bf16(aA[kc], ld_bfrag(Wr0T, DOUT, n, hi, kk), rac[t]);
      rac[t] = wmma_bf16(xA[kc], ld_bfrag(Wr1T, DOUT, n, hi, kk), rac[t]);
    }

  // Activations; park r (C-layout) in this wave's LDS tile for A-relayout.
  float* L = lds_r[wave];
#pragma unroll
  for (int t = 0; t < 4; ++t) {
    const int col = 16 * t + nl;
    const float bz = bz0[col] + bz1[col];
    const float br = br0[col] + br1[col];
#pragma unroll
    for (int r = 0; r < 8; ++r) {
      zac[t][r] = sigmoidf(zac[t][r] + bz);
      L[(r + 8 * hi) * 64 + col] = sigmoidf(rac[t][r] + br);
    }
  }
  __syncthreads();

  // rx fragment: read r at this lane's A-layout (m,K) slots, times x_enc.
  v16bf rx[2];
#pragma unroll
  for (int kc = 0; kc < 2; ++kc)
#pragma unroll
    for (int i = 0; i < 16; ++i) {
      const int K = kc * 32 + ((i & 8) ? 16 : 0) + (i & 7) + 8 * hi;
      rx[kc][i] = (__bf16)(L[m * 64 + K] * (float)xA[kc][i]);
    }

  v8f hac[4] = {};
#pragma unroll
  for (int t = 0; t < 4; ++t)
#pragma unroll
    for (int kc = 0; kc < 2; ++kc) {
      const int n = 16 * t + nl, kk = kc * 32;
      hac[t] = wmma_bf16(aA[kc], ld_bfrag(Wh0T, DOUT, n, hi, kk), hac[t]);
      hac[t] = wmma_bf16(rx[kc], ld_bfrag(Wh1T, DOUT, n, hi, kk), hac[t]);
    }

#pragma unroll
  for (int t = 0; t < 4; ++t) {
    const int col = 16 * t + nl;
    const float bh = bh0[col] + bh1[col];
#pragma unroll
    for (int r = 0; r < 8; ++r) {
      const int row = row0 + r + 8 * hi;
      float hv = fmaxf(mask[row] * (hac[t][r] + bh), 0.0f);
      const float zv = zac[t][r];
      const float xv = xe[(size_t)row * DOUT + col];
      out[(size_t)row * DOUT + col] = hv * zv + xv * (1.0f - zv);
    }
  }
}

// ---------------------------------------------------------------------------
extern "C" void kernel_launch(void* const* d_in, const int* in_sizes, int n_in,
                              void* d_out, int out_size, void* d_ws, size_t ws_size,
                              hipStream_t stream) {
  const float* mask     = (const float*)d_in[0];
  const float* adj      = (const float*)d_in[1];
  const float* x        = (const float*)d_in[2];
  const float* W_encode = (const float*)d_in[3];
  const float* W_z0     = (const float*)d_in[4];
  const float* W_z1     = (const float*)d_in[5];
  const float* W_r0     = (const float*)d_in[6];
  const float* W_r1     = (const float*)d_in[7];
  const float* W_h0     = (const float*)d_in[8];
  const float* W_h1     = (const float*)d_in[9];
  const float* b_encode = (const float*)d_in[10];
  const float* b_z0     = (const float*)d_in[11];
  const float* b_z1     = (const float*)d_in[12];
  const float* b_r0     = (const float*)d_in[13];
  const float* b_r1     = (const float*)d_in[14];
  const float* b_h0     = (const float*)d_in[15];
  const float* b_h1     = (const float*)d_in[16];

  char* ws = (char*)d_ws;
  __bf16* WeT  = (__bf16*)(ws + 0);
  __bf16* WT   = (__bf16*)(ws + 16384);
  __bf16* xT   = (__bf16*)(ws + 65536);
  float*  xe   = (float*)(ws + 2162688);
  float*  a_ws = (float*)(ws + 6356992);

  k_prep<<<32, 256, 0, stream>>>(W_encode, W_z0, W_z1, W_r0, W_r1, W_h0, W_h1,
                                 WeT, WT);
  k_encode<<<ROWS / 64, 128, 0, stream>>>(x, WeT, b_encode, xe, xT);

  const int n4 = ROWS * DOUT / 4;
  k_zero<<<(n4 + 255) / 256, 256, 0, stream>>>((float4*)a_ws, n4);

  dim3 g2(N_SZ / 64, B_SZ, KSPLIT);
  k_aggregate<<<g2, 128, 0, stream>>>(adj, xT, a_ws);

  k_gate<<<ROWS / 64, 128, 0, stream>>>(a_ws, xe, WT, mask,
                                        b_z0, b_z1, b_r0, b_r1, b_h0, b_h1,
                                        (float*)d_out);
}
